// CPC_2843268350719
// MI455X (gfx1250) — compile-verified, matrix operand order checked by reading, not run on hardware
//
#include <hip/hip_runtime.h>
#include <hip/hip_bf16.h>
#include <math.h>

// Problem constants (from reference)
#define B_   128
#define T_   512
#define W_   64
#define D_   128
#define C_   256
#define C3_  768
#define KS_  12
#define TP_  500            // T - K_STEPS
#define TEMP_INV 10.0f      // 1 / 0.1
#define EPS_ 1e-12f

typedef __attribute__((ext_vector_type(16))) __bf16 bf16x16;
typedef __attribute__((ext_vector_type(8)))  float  f32x8;

union AFrag { bf16x16 v; __bf16 e[16]; };

// Fragment loader for 16x32 bf16 A (lane = M row) or 32x16 bf16 B (lane = N col,
// source stored N-major so each "row" is contiguous in K).
// Per ISA 7.12.2 16-bit A layout: VGPR v holds K-pairs; lanes 16-31 are K+8.
// element i (0..15): K = ((i&8)<<1) + (i&7) + 8*(lane>>4)
// Per-lane bytes form two contiguous 16B chunks -> compiler emits global_load_b128 pairs.
__device__ __forceinline__ bf16x16 load_frag(const __bf16* base, size_t rowStride, int lane) {
  AFrag f;
  const __bf16* p = base + (size_t)(lane & 15) * rowStride + (((lane >> 4) & 1) << 3);
#pragma unroll
  for (int i = 0; i < 16; i += 2) {
    int kk = ((i & 8) << 1) + (i & 7);
    f.e[i]     = p[kk];
    f.e[i + 1] = p[kk + 1];
  }
  return f.v;
}

__device__ __forceinline__ f32x8 wmma_bf16(bf16x16 a, bf16x16 b, f32x8 c) {
  // D = A(16x32) * B(32x16) + C, f32 accumulate
  return __builtin_amdgcn_wmma_f32_16x16x32_bf16(false, a, false, b, (short)0, c, false, false);
}

// ---------------- init / weight conversion ----------------

__global__ void k_init(float* accum) {
  if (threadIdx.x < 2) accum[threadIdx.x] = 0.0f;
}

__global__ void k_convWi(const float* __restrict__ Wi, __bf16* __restrict__ WiT) {
  int i = blockIdx.x * blockDim.x + threadIdx.x;            // over C3_*D_
  if (i < C3_ * D_) {
    int n = i / D_, kk = i % D_;
    WiT[i] = (__bf16)Wi[(size_t)kk * C3_ + n];              // Wi (D,3C) -> (3C,D)
  }
}
__global__ void k_convWh(const float* __restrict__ Wh, __bf16* __restrict__ WhT) {
  int i = blockIdx.x * blockDim.x + threadIdx.x;            // over C3_*C_
  if (i < C3_ * C_) {
    int n = i / C_, kk = i % C_;
    WhT[i] = (__bf16)Wh[(size_t)kk * C3_ + n];              // Wh (C,3C) -> (3C,C)
  }
}
__global__ void k_convWk(const float* __restrict__ Wk, __bf16* __restrict__ WkT) {
  int i = blockIdx.x * blockDim.x + threadIdx.x;            // over KS_*D_*C_
  if (i < KS_ * D_ * C_) {
    int k = i / (D_ * C_);
    int r = i % (D_ * C_);
    int n = r / C_, cc = r % C_;
    WkT[i] = (__bf16)Wk[((size_t)k * C_ + cc) * D_ + n];    // Wk (K,C,D) -> (K,D,C)
  }
}

// ---------------- z = l2norm(rr @ We + be) ----------------

__global__ __launch_bounds__(128) void k_z(const float* __restrict__ rr,
                                           const float* __restrict__ We,
                                           const float* __restrict__ be,
                                           float* __restrict__ z_out,
                                           __bf16* __restrict__ zb) {
  const int row = blockIdx.x;          // b*T + t
  const int d   = threadIdx.x;         // 0..127
  __shared__ float rrs[W_];
  __shared__ float red[4];
  if (d < W_) rrs[d] = rr[(size_t)row * W_ + d];
  __syncthreads();
  float y = be[d];
#pragma unroll
  for (int w = 0; w < W_; ++w) y = fmaf(rrs[w], We[w * D_ + d], y);
  float s = y * y;
#pragma unroll
  for (int m = 16; m >= 1; m >>= 1) s += __shfl_xor(s, m, 32);
  if ((threadIdx.x & 31) == 0) red[threadIdx.x >> 5] = s;
  __syncthreads();
  float tot = red[0] + red[1] + red[2] + red[3];
  float inv = 1.0f / fmaxf(sqrtf(tot), EPS_);
  float zv  = y * inv;
  z_out[(size_t)row * D_ + d] = zv;
  zb[(size_t)row * D_ + d]    = (__bf16)zv;
}

// ---------------- GRU: 8 workgroups x 16 batch rows, t-loop inside ----------------
// Wave wv owns gate-triple columns [16wv, 16wv+16) of r/z/n.
// Accumulators: aR,aZ fuse x+h parts; aXN,aHN kept separate (n = tanh(xn + r*hn)).
// Weight fragments are deliberately re-fetched from L2 each step (see asm barrier):
// keeping 36 fragments live across the loop exceeds the 256-VGPR/wave budget and
// previously spilled every operand to scratch.

__global__ __launch_bounds__(512) void k_gru(const __bf16* __restrict__ zb,
                                             const __bf16* __restrict__ WiT,
                                             const __bf16* __restrict__ WhT,
                                             const float* __restrict__ bi,
                                             const float* __restrict__ bh,
                                             float* __restrict__ c_out,
                                             __bf16* __restrict__ cb) {
  const int bblk = blockIdx.x;         // batch rows 16*bblk .. +15
  const int tid  = threadIdx.x;
  const int wv   = tid >> 5;           // 0..15
  const int lane = tid & 31;
  const int nl   = lane & 15;
  const int half = lane >> 4;

  __shared__ float  hF[16][C_];        // fp32 h for gate blend
  __shared__ __bf16 hB[16][C_];        // bf16 h, A-operand source
  for (int i = tid; i < 16 * C_; i += 512) {
    hF[i >> 8][i & 255] = 0.0f;
    hB[i >> 8][i & 255] = (__bf16)0.0f;
  }
  __syncthreads();

  const int cc0 = wv * 16;             // gate-local column block
  const int col = cc0 + nl;            // this lane's gate column (0..255)
  const float bR  = bi[col] + bh[col];
  const float bZ  = bi[C_ + col] + bh[C_ + col];
  const float bXN = bi[2 * C_ + col];
  const float bHN = bh[2 * C_ + col];

  const size_t zRow = (size_t)T_ * D_;
  const __bf16* zbase0 = zb + (size_t)(16 * bblk) * zRow;
  // per-lane base of this lane's z row (prefetch target)
  const __bf16* zlane = zbase0 + (size_t)nl * zRow;

  for (int t = 0; t < T_; ++t) {
    // Compiler barrier: forbid hoisting the loop-invariant weight-fragment loads
    // (they would spill to scratch); re-read them from L2 every step instead.
    asm volatile("" ::: "memory");

    // prefetch next step's z row for this lane (global_prefetch_b8)
    if (t + 1 < T_) __builtin_prefetch(zlane + (size_t)(t + 1) * D_, 0, 0);

    f32x8 aR, aZ, aXN, aHN;
#pragma unroll
    for (int e = 0; e < 8; ++e) { aR[e] = 0.f; aZ[e] = 0.f; aXN[e] = 0.f; aHN[e] = 0.f; }

    // input projection: z_t(16x128) @ Wi(128x768), our 3 gate tiles
#pragma unroll
    for (int ks = 0; ks < D_ / 32; ++ks) {
      bf16x16 a  = load_frag(zbase0 + (size_t)t * D_ + ks * 32, zRow, lane);
      bf16x16 br = load_frag(WiT + (size_t)cc0 * D_            + ks * 32, D_, lane);
      bf16x16 bz = load_frag(WiT + (size_t)(C_ + cc0) * D_     + ks * 32, D_, lane);
      bf16x16 bn = load_frag(WiT + (size_t)(2 * C_ + cc0) * D_ + ks * 32, D_, lane);
      aR  = wmma_bf16(a, br, aR);
      aZ  = wmma_bf16(a, bz, aZ);
      aXN = wmma_bf16(a, bn, aXN);
    }
    // recurrent: h(16x256) @ Wh(256x768)
#pragma unroll
    for (int ks = 0; ks < C_ / 32; ++ks) {
      bf16x16 a  = load_frag(&hB[0][0] + ks * 32, C_, lane);
      bf16x16 br = load_frag(WhT + (size_t)cc0 * C_            + ks * 32, C_, lane);
      bf16x16 bz = load_frag(WhT + (size_t)(C_ + cc0) * C_     + ks * 32, C_, lane);
      bf16x16 bn = load_frag(WhT + (size_t)(2 * C_ + cc0) * C_ + ks * 32, C_, lane);
      aR  = wmma_bf16(a, br, aR);
      aZ  = wmma_bf16(a, bz, aZ);
      aHN = wmma_bf16(a, bn, aHN);
    }
    __syncthreads();   // everyone done reading hB

    float hnew[8];
#pragma unroll
    for (int v = 0; v < 8; ++v) {
      int m = v + 8 * half;            // local batch row (C/D layout)
      float r  = 1.0f / (1.0f + __expf(-(aR[v] + bR)));
      float zg = 1.0f / (1.0f + __expf(-(aZ[v] + bZ)));
      float n  = tanhf(aXN[v] + bXN + r * (aHN[v] + bHN));
      hnew[v]  = (1.0f - zg) * n + zg * hF[m][col];
    }
#pragma unroll
    for (int v = 0; v < 8; ++v) {
      int m = v + 8 * half;
      hF[m][col] = hnew[v];
      hB[m][col] = (__bf16)hnew[v];
      size_t gi = ((size_t)(16 * bblk + m) * T_ + t) * C_ + col;
      c_out[gi] = hnew[v];
      cb[gi]    = (__bf16)hnew[v];
    }
    __syncthreads();   // h_new visible before next step's GEMM
  }
}

// ---------------- fused pred -> l2norm -> logits -> lse/diag/argmax ----------------
// grid = (Tp, K); block = 8 waves; wave wv owns pred/logits rows 16wv..+15.

__global__ __launch_bounds__(256) void k_pred(const __bf16* __restrict__ cb,
                                              const __bf16* __restrict__ zb,
                                              const __bf16* __restrict__ WkT,
                                              const float* __restrict__ Wkb,
                                              float* __restrict__ accum) {
  const int t    = blockIdx.x;   // 0..Tp-1
  const int k    = blockIdx.y;   // 0..K-1
  const int tid  = threadIdx.x;
  const int wv   = tid >> 5;     // 0..7 row block
  const int lane = tid & 31;
  const int nl   = lane & 15;
  const int half = lane >> 4;

  __shared__ __bf16 predB[128][128];   // normalized pred, row-major (32KB)

  const size_t cRow = (size_t)T_ * C_;
  const __bf16* abase  = cb + (size_t)(16 * wv) * cRow + (size_t)t * C_;
  const __bf16* wkbase = WkT + (size_t)k * D_ * C_;

  // GEMM1: pred(128x128) = c_t(128x256) @ Wk^T-stored(256x128)
  f32x8 acc[8];
#pragma unroll
  for (int j = 0; j < 8; ++j)
#pragma unroll
    for (int e = 0; e < 8; ++e) acc[j][e] = 0.0f;

#pragma unroll
  for (int ks = 0; ks < C_ / 32; ++ks) {
    bf16x16 a = load_frag(abase + ks * 32, cRow, lane);
#pragma unroll
    for (int j = 0; j < 8; ++j) {
      bf16x16 b = load_frag(wkbase + (size_t)(16 * j) * C_ + ks * 32, C_, lane);
      acc[j] = wmma_bf16(a, b, acc[j]);
    }
  }

  // bias (in place) + row sum-of-squares (rows live in 16-lane halves)
  float ss[8];
#pragma unroll
  for (int v = 0; v < 8; ++v) ss[v] = 0.0f;
#pragma unroll
  for (int j = 0; j < 8; ++j) {
    float wb = Wkb[k * D_ + 16 * j + nl];
#pragma unroll
    for (int v = 0; v < 8; ++v) {
      float x = acc[j][v] + wb;
      acc[j][v] = x;
      ss[v]     = fmaf(x, x, ss[v]);
    }
  }
#pragma unroll
  for (int v = 0; v < 8; ++v) {
    float s = ss[v];
    s += __shfl_xor(s, 1, 32); s += __shfl_xor(s, 2, 32);
    s += __shfl_xor(s, 4, 32); s += __shfl_xor(s, 8, 32);
    ss[v] = 1.0f / fmaxf(sqrtf(s), EPS_);
  }
#pragma unroll
  for (int j = 0; j < 8; ++j)
#pragma unroll
    for (int v = 0; v < 8; ++v)
      predB[16 * wv + v + 8 * half][16 * j + nl] = (__bf16)(acc[j][v] * ss[v]);
  __syncthreads();

  // GEMM2: logits(128x128) = pred @ zf^T ; zf rows = z[c, t+k+1, :]
  f32x8 lg[8];
#pragma unroll
  for (int j = 0; j < 8; ++j)
#pragma unroll
    for (int e = 0; e < 8; ++e) lg[j][e] = 0.0f;

  const size_t zRow = (size_t)T_ * D_;
  const __bf16* zfbase = zb + (size_t)(t + k + 1) * D_;
#pragma unroll
  for (int ks = 0; ks < D_ / 32; ++ks) {
    bf16x16 a = load_frag(&predB[16 * wv][0] + ks * 32, 128, lane);
#pragma unroll
    for (int j = 0; j < 8; ++j) {
      bf16x16 b = load_frag(zfbase + (size_t)(16 * j) * zRow + ks * 32, zRow, lane);
      lg[j] = wmma_bf16(a, b, lg[j]);
    }
  }

  // per-row lse, diag, argmax; accumulate loss/accuracy
  float lossLocal = 0.0f, corrLocal = 0.0f;
#pragma unroll
  for (int v = 0; v < 8; ++v) {
    float mx = -3.4e38f, diag = 0.0f;
    int   am = 0x7fffffff;
#pragma unroll
    for (int j = 0; j < 8; ++j) {
      float x = lg[j][v] * TEMP_INV;
      int   c = 16 * j + nl;
      if (x > mx || (x == mx && c < am)) { mx = x; am = c; }
      if (j == wv) diag = x;             // diag col lives in tile j == wv
    }
#pragma unroll
    for (int msk = 1; msk <= 8; msk <<= 1) {
      float ox = __shfl_xor(mx, msk, 32);
      int   oa = __shfl_xor(am, msk, 32);
      if (ox > mx || (ox == mx && oa < am)) { mx = ox; am = oa; }
    }
    float se = 0.0f;
#pragma unroll
    for (int j = 0; j < 8; ++j) se += __expf(lg[j][v] * TEMP_INV - mx);
#pragma unroll
    for (int msk = 1; msk <= 8; msk <<= 1) se += __shfl_xor(se, msk, 32);
    float lse = mx + __logf(se);

    // diag lane for row m=v+8*half: (half==0 && nl==v) or (half==1 && nl==v+8)
    bool dl = (half == 0) ? (nl == v) : (nl == v + 8);
    if (dl) {
      int rb = 16 * wv + v + 8 * half;   // global row index b
      lossLocal += lse - diag;
      if (am == rb) corrLocal += 1.0f;
    }
  }
#pragma unroll
  for (int m = 16; m >= 1; m >>= 1) {
    lossLocal += __shfl_xor(lossLocal, m, 32);
    corrLocal += __shfl_xor(corrLocal, m, 32);
  }
  if (lane == 0) {
    atomicAdd(&accum[0], lossLocal);
    atomicAdd(&accum[1], corrLocal);
  }
}

__global__ void k_fin(const float* __restrict__ accum, float* __restrict__ out) {
  if (threadIdx.x == 0) {
    const float denom = (float)(KS_ * TP_ * B_);
    out[0] = accum[0] / denom;
    out[1] = 100.0f * accum[1] / denom;
  }
}

// ---------------- launch ----------------

extern "C" void kernel_launch(void* const* d_in, const int* in_sizes, int n_in,
                              void* d_out, int out_size, void* d_ws, size_t ws_size,
                              hipStream_t stream) {
  (void)in_sizes; (void)n_in; (void)out_size; (void)ws_size;

  const float* rr  = (const float*)d_in[0];
  const float* We  = (const float*)d_in[1];
  const float* be  = (const float*)d_in[2];
  const float* Wi  = (const float*)d_in[3];
  const float* Wh  = (const float*)d_in[4];
  const float* bi  = (const float*)d_in[5];
  const float* bh  = (const float*)d_in[6];
  const float* Wkw = (const float*)d_in[7];
  const float* Wkb = (const float*)d_in[8];

  float* out   = (float*)d_out;
  float* z_out = out + 2;                                   // (B,T,D) fp32
  float* c_out = z_out + (size_t)B_ * T_ * D_;              // (B,T,C) fp32

  char* ws = (char*)d_ws;
  __bf16* zb  = (__bf16*)ws;  ws += (size_t)B_ * T_ * D_ * 2;
  __bf16* cbf = (__bf16*)ws;  ws += (size_t)B_ * T_ * C_ * 2;
  __bf16* WiT = (__bf16*)ws;  ws += (size_t)C3_ * D_ * 2;
  __bf16* WhT = (__bf16*)ws;  ws += (size_t)C3_ * C_ * 2;
  __bf16* WkT = (__bf16*)ws;  ws += (size_t)KS_ * D_ * C_ * 2;
  float*  accum = (float*)ws;                               // 2 floats

  k_init<<<1, 32, 0, stream>>>(accum);
  k_convWi<<<(C3_ * D_ + 255) / 256, 256, 0, stream>>>(Wi, WiT);
  k_convWh<<<(C3_ * C_ + 255) / 256, 256, 0, stream>>>(Wh, WhT);
  k_convWk<<<(KS_ * D_ * C_ + 255) / 256, 256, 0, stream>>>(Wkw, WkT);
  k_z<<<B_ * T_, D_, 0, stream>>>(rr, We, be, z_out, zb);
  k_gru<<<B_ / 16, 512, 0, stream>>>(zb, WiT, WhT, bi, bh, c_out, cbf);
  dim3 gp(TP_, KS_);
  k_pred<<<gp, 256, 0, stream>>>(cbf, zb, WkT, Wkb, accum);
  k_fin<<<1, 32, 0, stream>>>(accum, out);
}
